// LeakyRNN_30872224923982
// MI455X (gfx1250) — matrix operand order repaired and checked by reading, not compiled, not run on hardware
//
#include <hip/hip_runtime.h>
#include <hip/hip_bf16.h>
#include <math.h>

// Problem constants (from reference)
#define Bn 64
#define Tn 512
#define In 256
#define Hn 1024
#define NOISE 0.1f

typedef __attribute__((ext_vector_type(16))) _Float16 v16h;
typedef __attribute__((ext_vector_type(8)))  _Float16 v8h;
typedef __attribute__((ext_vector_type(8)))  float    v8f;

// ---------------------------------------------------------------------------
// Prep 1: repack W (H x H fp32, row-major, out[b,j]=sum_k h[b,k]*W[k,j]) into
// WMMA-B-ready f16 tiles. Tile (kt,nt) covers K=[kt*32,kt*32+32), N=[nt*16,..).
// Per-lane layout: lane holds column n = nt*16 + (lane&15); element j holds
// K = kt*32 + (lane>>4)*16 + j  -> 16 contiguous f16 (32B) per lane.
// ---------------------------------------------------------------------------
__global__ void pack_W_kernel(const float* __restrict__ W, _Float16* __restrict__ Wp) {
    int gid  = blockIdx.x * blockDim.x + threadIdx.x;   // 2048 tiles * 32 lanes
    int lane = gid & 31;
    int tile = gid >> 5;            // 0..2047
    if (tile >= (Hn / 32) * (Hn / 16)) return;
    int kt = tile >> 6;             // 0..31
    int nt = tile & 63;             // 0..63
    int n  = nt * 16 + (lane & 15);
    int kb = kt * 32 + (lane >> 4) * 16;
    _Float16* dst = Wp + (size_t)tile * 512 + lane * 16;
#pragma unroll
    for (int j = 0; j < 16; ++j)
        dst[j] = (_Float16)W[(size_t)(kb + j) * Hn + n];
}

// ---------------------------------------------------------------------------
// Prep 2: repack in_w (H x I fp32, row-major; effective B[k][n] = in_w[n][k])
// into the same f16 B layout. 8 k-tiles (K=256) x 64 n-tiles.
// ---------------------------------------------------------------------------
__global__ void pack_I_kernel(const float* __restrict__ in_w, _Float16* __restrict__ Ip) {
    int gid  = blockIdx.x * blockDim.x + threadIdx.x;   // 512 tiles * 32 lanes
    int lane = gid & 31;
    int tile = gid >> 5;            // 0..511
    if (tile >= (In / 32) * (Hn / 16)) return;
    int kt = tile >> 6;             // 0..7
    int nt = tile & 63;
    int n  = nt * 16 + (lane & 15);
    int kb = kt * 32 + (lane >> 4) * 16;
    _Float16* dst = Ip + (size_t)tile * 512 + lane * 16;
#pragma unroll
    for (int j = 0; j < 16; ++j)
        dst[j] = (_Float16)in_w[(size_t)n * In + kb + j];
}

// ---------------------------------------------------------------------------
// Prep 3: alpha = DT / (sigmoid(taus)*90 + 10); zero both h double-buffers.
// ---------------------------------------------------------------------------
__global__ void prep_state_kernel(const float* __restrict__ taus, float* __restrict__ alpha,
                                  float* __restrict__ hA32, float* __restrict__ hB32,
                                  _Float16* __restrict__ hA16, _Float16* __restrict__ hB16) {
    int gid = blockIdx.x * blockDim.x + threadIdx.x;
    if (gid < Bn * Hn) {
        hA32[gid] = 0.0f; hB32[gid] = 0.0f;
        hA16[gid] = (_Float16)0.0f; hB16[gid] = (_Float16)0.0f;
    }
    if (gid < Hn) {
        float s   = 1.0f / (1.0f + __expf(-taus[gid]));
        float tau = s * 90.0f + 10.0f;
        alpha[gid] = 10.0f / tau;     // DT / tau
    }
}

// ---------------------------------------------------------------------------
// One recurrence step:  h' = (1-a)*h + a*tanh(x_t@in_w^T + b + h@W + 0.1*eps)
// Grid: 256 blocks (4 M-tiles x 64 N-tiles), 128 threads = 4 wave32.
// Each block owns one 16x16 output tile; the K=1280 reduction (256 x-proj +
// 1024 recurrent) is split across the 4 waves (10 k-chunks of 32 each), then
// reduced through LDS. Wave 0 applies the leaky-tanh update and writes out.
// ---------------------------------------------------------------------------
__global__ void __launch_bounds__(128)
leaky_rnn_step_kernel(const float*    __restrict__ x,      // [B,T,I]
                      const float*    __restrict__ eps,    // [B,T,H]
                      const float*    __restrict__ in_b,   // [H]
                      const _Float16* __restrict__ Wp,     // packed W
                      const _Float16* __restrict__ Ip,     // packed in_w^T
                      const float*    __restrict__ alpha,  // [H]
                      const float*    __restrict__ h_in32, // [B,H]
                      const _Float16* __restrict__ h_in16, // [B,H]
                      float*          __restrict__ h_out32,
                      _Float16*       __restrict__ h_out16,
                      float*          __restrict__ out,    // [B,T,H]
                      int t) {
    const int lane  = threadIdx.x & 31;
    const int w     = threadIdx.x >> 5;     // wave id 0..3
    const int mt    = blockIdx.x >> 6;      // 0..3   (M tile; M == batch)
    const int nt    = blockIdx.x & 63;      // 0..63  (N tile)
    const int khalf = lane >> 4;            // A/B layout half-select
    const int m     = mt * 16 + (lane & 15);

    v8f acc = {};

    // ---- input-projection slice: k-chunks 2w, 2w+1 over K=256 ----
    const float* xrow = x + ((size_t)m * Tn + t) * In;
#pragma unroll
    for (int c = 0; c < 2; ++c) {
        int kc = 2 * w + c;
        int kb = kc * 32;
        v8f f0 = *(const v8f*)(xrow + kb + khalf * 16);
        v8f f1 = *(const v8f*)(xrow + kb + khalf * 16 + 8);
        v8h lo = __builtin_convertvector(f0, v8h);
        v8h hi = __builtin_convertvector(f1, v8h);
        v16h a = __builtin_shufflevector(lo, hi, 0,1,2,3,4,5,6,7,8,9,10,11,12,13,14,15);
        v16h bm = *(const v16h*)(Ip + ((size_t)(kc * 64 + nt)) * 512 + lane * 16);
        acc = __builtin_amdgcn_wmma_f32_16x16x32_f16(false, a, false, bm,
                                                     (short)0, acc, false, false);
    }

    // ---- recurrent slice: k-chunks 8w .. 8w+7 over K=1024 ----
    const _Float16* hrow = h_in16 + (size_t)m * Hn;
#pragma unroll
    for (int c = 0; c < 8; ++c) {
        int kc = 8 * w + c;
        int kb = kc * 32;
        v8h lo = *(const v8h*)(hrow + kb + khalf * 16);
        v8h hi = *(const v8h*)(hrow + kb + khalf * 16 + 8);
        v16h a = __builtin_shufflevector(lo, hi, 0,1,2,3,4,5,6,7,8,9,10,11,12,13,14,15);
        v16h bm = *(const v16h*)(Wp + ((size_t)(kc * 64 + nt)) * 512 + lane * 16);
        acc = __builtin_amdgcn_wmma_f32_16x16x32_f16(false, a, false, bm,
                                                     (short)0, acc, false, false);
    }

    // ---- cross-wave K reduction through LDS ----
    __shared__ v8f red[4][32];
    red[w][lane] = acc;
    __syncthreads();

    if (w == 0) {
        v8f s = red[0][lane] + red[1][lane] + red[2][lane] + red[3][lane];
        const int n     = nt * 16 + (lane & 15);
        const int mbase = mt * 16 + khalf * 8;   // C layout: VGPR r -> M = khalf*8 + r
        const float bn    = in_b[n];
        const float al    = alpha[n];
        const float one_m = 1.0f - al;
#pragma unroll
        for (int r = 0; r < 8; ++r) {
            int    b   = mbase + r;
            size_t idx = ((size_t)b * Tn + t) * Hn + n;
            float  pre = s[r] + bn + NOISE * eps[idx];
            float  nh  = one_m * h_in32[b * Hn + n] + al * tanhf(pre);
            out[idx]            = nh;
            h_out32[b * Hn + n] = nh;
            h_out16[b * Hn + n] = (_Float16)nh;
        }
    }
}

// ---------------------------------------------------------------------------
extern "C" void kernel_launch(void* const* d_in, const int* in_sizes, int n_in,
                              void* d_out, int out_size, void* d_ws, size_t ws_size,
                              hipStream_t stream) {
    const float* x    = (const float*)d_in[0];  // [B,T,I]
    const float* eps  = (const float*)d_in[1];  // [B,T,H]
    const float* in_w = (const float*)d_in[2];  // [H,I]
    const float* in_b = (const float*)d_in[3];  // [H]
    const float* W    = (const float*)d_in[4];  // [H,H]
    const float* taus = (const float*)d_in[5];  // [H]
    float*       out  = (float*)d_out;          // [B,T,H]

    // Workspace layout (~3.3 MB total)
    char* ws = (char*)d_ws;
    _Float16* Wp    = (_Float16*)(ws);                         // 2 MB
    _Float16* Ip    = (_Float16*)(ws + 2097152);               // 512 KB
    float*    alpha = (float*)   (ws + 2621440);               // 4 KB
    float*    hA32  = (float*)   (ws + 2625536);               // 256 KB
    float*    hB32  = (float*)   (ws + 2887680);               // 256 KB
    _Float16* hA16  = (_Float16*)(ws + 3149824);               // 128 KB
    _Float16* hB16  = (_Float16*)(ws + 3280896);               // 128 KB

    // One-time (per call) prep: pack operands, compute alpha, zero state.
    pack_W_kernel<<<256, 256, 0, stream>>>(W, Wp);             // 65536 threads
    pack_I_kernel<<<64, 256, 0, stream>>>(in_w, Ip);           // 16384 threads
    prep_state_kernel<<<256, 256, 0, stream>>>(taus, alpha, hA32, hB32, hA16, hB16);

    // Sequential recurrence: 512 step launches, double-buffered h state.
    for (int t = 0; t < Tn; ++t) {
        const float*    hin32  = (t & 1) ? hB32 : hA32;
        const _Float16* hin16  = (t & 1) ? hB16 : hA16;
        float*          hout32 = (t & 1) ? hA32 : hB32;
        _Float16*       hout16 = (t & 1) ? hA16 : hB16;
        leaky_rnn_step_kernel<<<256, 128, 0, stream>>>(
            x, eps, in_b, Wp, Ip, alpha, hin32, hin16, hout32, hout16, out, t);
    }
}
